// Selection_model_21371757265330
// MI455X (gfx1250) — matrix operand order repaired and checked by reading, not compiled, not run on hardware
//
#include <hip/hip_runtime.h>

typedef __bf16 bf16;
typedef __attribute__((ext_vector_type(16))) __bf16 v16bf;
typedef __attribute__((ext_vector_type(8)))  float  v8f;
typedef __attribute__((ext_vector_type(8)))  unsigned v8u;

#define DM    128   // embedding dim
#define HN    8     // heads
#define DK    16    // head dim
#define FFD   512   // ff hidden
#define BATCH 32

__device__ __forceinline__ float geluf(float x) {
    return 0.5f * x * (1.0f + erff(x * 0.70710678118654752f));
}

// ---------------------------------------------------------------------------
// f32 [Din][Dout] -> bf16 transposed [Dout][Din]
// ---------------------------------------------------------------------------
__global__ void k_cvt_t(const float* __restrict__ s, bf16* __restrict__ d,
                        int Din, int Dout) {
    int i = blockIdx.x * blockDim.x + threadIdx.x;
    if (i >= Din * Dout) return;
    int k = i / Dout, c = i % Dout;
    d[(size_t)c * Din + k] = (bf16)s[i];
}

// ---------------------------------------------------------------------------
// embedding: data[B*N,2] @ emb_W[2,128] + emb_b
// ---------------------------------------------------------------------------
__global__ void k_embed(const float* __restrict__ data, const float* __restrict__ W,
                        const float* __restrict__ b,
                        float* __restrict__ X, bf16* __restrict__ Xb, int R) {
    int i = blockIdx.x * blockDim.x + threadIdx.x;
    if (i >= R * DM) return;
    int r = i >> 7, c = i & (DM - 1);
    float v = data[r * 2 + 0] * W[c] + data[r * 2 + 1] * W[DM + c] + b[c];
    X[i]  = v;
    Xb[i] = (bf16)v;
}

// ---------------------------------------------------------------------------
// generic bf16 WMMA GEMM, weights pre-transposed WT[Dout][Din].
// One wave computes a 16x32 tile (two accumulators share the A fragment).
// grid = (R/16, Dout/32). All fragment loads are dword-wide.
// OBFT: additionally store bf16 output per-batch transposed [b][col][n] (for V).
// ---------------------------------------------------------------------------
template<bool BIAS, bool RESID, bool RELU, bool O32, bool OBF, bool OBFT>
__global__ void k_gemm(const bf16* __restrict__ A, const bf16* __restrict__ WT,
                       const float* __restrict__ bias, const float* __restrict__ resid,
                       float* __restrict__ out32, bf16* __restrict__ outbf,
                       bf16* __restrict__ outbfT,
                       int Din, int Dout, int logN) {
    int lane = threadIdx.x & 31;
    int g = lane >> 4, ln = lane & 15;
    int mt = blockIdx.x, nt = blockIdx.y;
    const bf16* Ar  = A  + (size_t)(mt * 16 + ln) * Din;        // A row for this lane
    const bf16* Wr0 = WT + (size_t)(nt * 32 + ln) * Din;        // WT rows (= B cols)
    const bf16* Wr1 = Wr0 + (size_t)16 * Din;
    v8f c0 = {}, c1 = {};
    for (int k0 = 0; k0 < Din; k0 += 32) {
        v8u au, bu0, bu1;
#pragma unroll
        for (int j = 0; j < 8; ++j) {
            int ka = k0 + (g << 4) + 2 * j;
            au[j] = *(const unsigned*)(Ar + ka);
            int kb = k0 + (g << 4) + 2 * j;
            bu0[j] = *(const unsigned*)(Wr0 + kb);
            bu1[j] = *(const unsigned*)(Wr1 + kb);
        }
        v16bf a  = __builtin_bit_cast(v16bf, au);
        v16bf b0 = __builtin_bit_cast(v16bf, bu0);
        v16bf b1 = __builtin_bit_cast(v16bf, bu1);
        c0 = __builtin_amdgcn_wmma_f32_16x16x32_bf16(false, a, false, b0, (short)0, c0, false, false);
        c1 = __builtin_amdgcn_wmma_f32_16x16x32_bf16(false, a, false, b1, (short)0, c1, false, false);
    }
#pragma unroll
    for (int t = 0; t < 2; ++t) {
        int col = nt * 32 + t * 16 + ln;
        float bval = 0.0f;
        if constexpr (BIAS) bval = bias[col];
#pragma unroll
        for (int i = 0; i < 8; ++i) {
            int r = mt * 16 + i + g * 8;
            float v = (t ? c1[i] : c0[i]) + bval;
            if constexpr (RESID) v += resid[(size_t)r * Dout + col];
            if constexpr (RELU)  v = fmaxf(v, 0.0f);
            if constexpr (O32)   out32[(size_t)r * Dout + col] = v;
            if constexpr (OBF)   outbf[(size_t)r * Dout + col] = (bf16)v;
            if constexpr (OBFT) {
                int bb = r >> logN, nl = r & ((1 << logN) - 1);
                outbfT[((size_t)bb * DM + col) << logN | nl] = (bf16)v;
            }
        }
    }
}

// ---------------------------------------------------------------------------
// fused attention for one (b, h, 16-query tile): softmax(Q K^T * scale + mask) V
// Q/K bf16 [B*N][128]; VT bf16 [B][128][N] (per-batch transposed); O bf16 [B*N][128]
// LDS: 16*N floats of scores.
// ---------------------------------------------------------------------------
__global__ void k_attn(const bf16* __restrict__ Q, const bf16* __restrict__ K,
                       const bf16* __restrict__ VT, const float* __restrict__ mask,
                       bf16* __restrict__ O, int N) {
    extern __shared__ float S[];           // [16][N]
    int lane = threadIdx.x & 31;
    int g = lane >> 4, ln = lane & 15;
    int qt = blockIdx.x, h = blockIdx.y, b = blockIdx.z;
    int hoff = h * DK;
    const bf16* Qb  = Q  + (size_t)b * N * DM;
    const bf16* Kb  = K  + (size_t)b * N * DM;
    const bf16* VTc = VT + ((size_t)b * DM + hoff + ln) * N;   // V^T row for lane's dk
    const float scale = 0.25f;             // 1/sqrt(16)

    // Q A-fragment: rows = query rows, K dim = head dim padded 16->32 (upper half zero)
    int qrow = qt * 16 + ln;
    v8u qu = {};
#pragma unroll
    for (int j = 0; j < 4; ++j)
        qu[j] = *(const unsigned*)(Qb + (size_t)qrow * DM + hoff + (g << 3) + 2 * j);
    v16bf aq = __builtin_bit_cast(v16bf, qu);

    // pass 1: scores into LDS
    for (int j = 0; j < N / 16; ++j) {
        int key = j * 16 + ln;
        v8u ku = {};
        {                                  // K rows 16..31 of the B fragment are zero pad
#pragma unroll
            for (int jj = 0; jj < 4; ++jj)
                ku[jj] = *(const unsigned*)(Kb + (size_t)key * DM + hoff + (g << 3) + 2 * jj);
        }
        v16bf bk = __builtin_bit_cast(v16bf, ku);
        v8f c = {};
        c = __builtin_amdgcn_wmma_f32_16x16x32_bf16(false, aq, false, bk, (short)0, c, false, false);
        float mcol = mask ? mask[(size_t)b * N + key] : 0.0f;
#pragma unroll
        for (int i = 0; i < 8; ++i)
            S[(i + g * 8) * N + j * 16 + ln] = c[i] * scale + mcol;
    }
    __syncthreads();

    // softmax per row: two lanes per row (lane ln and ln+16 split the key range)
    {
        int half = N >> 1;
        int base = g * half;
        float* row = S + ln * N;
        float m = -__builtin_inff();
        for (int n = 0; n < half; ++n) m = fmaxf(m, row[base + n]);
        m = fmaxf(m, __shfl_xor(m, 16));
        float s = 0.0f;
        for (int n = 0; n < half; ++n) {
            float e = __expf(row[base + n] - m);
            row[base + n] = e;
            s += e;
        }
        s += __shfl_xor(s, 16);
        float inv = 1.0f / s;
        for (int n = 0; n < half; ++n) row[base + n] *= inv;
    }
    __syncthreads();

    // pass 2: O = P @ V, K dim = keys, 32 per WMMA; V fragment is dword loads from VT
    v8f o = {};
    for (int j = 0; j < N / 32; ++j) {
        v16bf ap;
#pragma unroll
        for (int e = 0; e < 16; ++e) {
            int ka = j * 32 + (g << 4) + e;
            ap[e] = (bf16)S[ln * N + ka];
        }
        v8u vu;
#pragma unroll
        for (int jj = 0; jj < 8; ++jj)
            vu[jj] = *(const unsigned*)(VTc + j * 32 + (g << 4) + 2 * jj);
        v16bf bv = __builtin_bit_cast(v16bf, vu);
        o = __builtin_amdgcn_wmma_f32_16x16x32_bf16(false, ap, false, bv, (short)0, o, false, false);
    }
#pragma unroll
    for (int i = 0; i < 8; ++i) {
        int r = qt * 16 + i + g * 8;
        O[((size_t)b * N + r) * DM + hoff + ln] = (bf16)o[i];
    }
}

// ---------------------------------------------------------------------------
// instance norm over node axis per (batch, channel); biased var, eps=1e-5
// ---------------------------------------------------------------------------
__global__ void k_inorm(const float* __restrict__ in, const float* __restrict__ gam,
                        const float* __restrict__ bet,
                        float* __restrict__ out32, bf16* __restrict__ outbf, int N) {
    int idx = blockIdx.x * blockDim.x + threadIdx.x;
    if (idx >= BATCH * DM) return;
    int b = idx / DM, c = idx % DM;
    const float* p = in + (size_t)b * N * DM + c;
    float s = 0.0f, s2 = 0.0f;
    for (int n = 0; n < N; ++n) { float v = p[(size_t)n * DM]; s += v; s2 += v * v; }
    float mean = s / (float)N;
    float var  = s2 / (float)N - mean * mean;
    float rs   = rsqrtf(var + 1e-5f);
    float gg = gam[c], bb = bet[c];
    float* q = out32 + (size_t)b * N * DM + c;
    bf16*  qb = outbf ? outbf + (size_t)b * N * DM + c : nullptr;
    for (int n = 0; n < N; ++n) {
        float v = (p[(size_t)n * DM] - mean) * rs * gg + bb;
        q[(size_t)n * DM] = v;
        if (qb) qb[(size_t)n * DM] = (bf16)v;
    }
}

// ---------------------------------------------------------------------------
// scores = tanh(se @ pw + pb) + mask
// ---------------------------------------------------------------------------
__global__ void k_score(const float* __restrict__ se, const float* __restrict__ pw,
                        const float* __restrict__ pb, const float* __restrict__ mask,
                        float* __restrict__ sc, int R) {
    int idx = blockIdx.x * blockDim.x + threadIdx.x;
    if (idx >= R) return;
    const float* x = se + (size_t)idx * DM;
    float s = 0.0f;
    for (int c = 0; c < DM; ++c) s += x[c] * pw[c];
    s = tanhf(s + pb[0]);
    if (mask) s += mask[idx];
    sc[idx] = s;
}

// ---------------------------------------------------------------------------
// top-k (descending, lowest-index ties) + gather + add score; one wave per batch
// ---------------------------------------------------------------------------
__global__ void k_topk(const float* __restrict__ scores, const float* __restrict__ Xin,
                       float* __restrict__ Xout, bf16* __restrict__ Xoutbf,
                       int N, int kk) {
    __shared__ float sc[512];
    int b = blockIdx.x;
    int lane = threadIdx.x & 31;
    for (int n = lane; n < N; n += 32) sc[n] = scores[(size_t)b * N + n];
    __syncthreads();
    for (int i = 0; i < kk; ++i) {
        float m = -__builtin_inff();
        int mi = -1;
        for (int n = lane; n < N; n += 32) {
            float v = sc[n];
            if (v > m) { m = v; mi = n; }
        }
        for (int off = 16; off; off >>= 1) {
            float om = __shfl_xor(m, off);
            int  omi = __shfl_xor(mi, off);
            if (om > m || (om == m && omi >= 0 && (mi < 0 || omi < mi))) { m = om; mi = omi; }
        }
        if (lane == 0) sc[mi] = -__builtin_inff();
        for (int c = lane; c < DM; c += 32) {
            float v = Xin[((size_t)b * N + mi) * DM + c] + m;
            Xout[((size_t)b * kk + i) * DM + c]   = v;
            Xoutbf[((size_t)b * kk + i) * DM + c] = (bf16)v;
        }
        __syncthreads();
    }
}

// ---------------------------------------------------------------------------
// final pooling: GELU(concat(mean over nodes, max over nodes)) -> [B, 2*DM]
// ---------------------------------------------------------------------------
__global__ void k_pool(const float* __restrict__ X, float* __restrict__ out, int N) {
    int idx = blockIdx.x * blockDim.x + threadIdx.x;
    if (idx >= BATCH * DM) return;
    int b = idx >> 7, c = idx & (DM - 1);
    const float* p = X + (size_t)b * N * DM + c;
    float s = 0.0f, mx = -__builtin_inff();
    for (int n = 0; n < N; ++n) { float v = p[(size_t)n * DM]; s += v; mx = fmaxf(mx, v); }
    out[b * (2 * DM) + c]      = geluf(s / (float)N);
    out[b * (2 * DM) + DM + c] = geluf(mx);
}

// ---------------------------------------------------------------------------
// host orchestration
// ---------------------------------------------------------------------------
extern "C" void kernel_launch(void* const* d_in, const int* in_sizes, int n_in,
                              void* d_out, int out_size, void* d_ws, size_t ws_size,
                              hipStream_t stream) {
    (void)in_sizes; (void)n_in; (void)out_size; (void)ws_size;

    const float* data  = (const float*)d_in[0];
    const float* mask0 = (const float*)d_in[1];
    const float* embW  = (const float*)d_in[2];
    const float* embB  = (const float*)d_in[3];
    const int LBASE = 4, NLY = 12, NPARAM = 13, PBASE = 4 + 12 * 13;

    // workspace carve-out
    char* wsp = (char*)d_ws;
    size_t off = 0;
    auto take = [&](size_t bytes) -> char* {
        char* p = wsp + off;
        off += (bytes + 255) & ~(size_t)255;
        return p;
    };
    float* XA32 = (float*)take((size_t)BATCH * 512 * DM * 4);
    bf16*  XAbf = (bf16*) take((size_t)BATCH * 512 * DM * 2);
    float* XB32 = (float*)take((size_t)BATCH * 256 * DM * 4);
    bf16*  XBbf = (bf16*) take((size_t)BATCH * 256 * DM * 2);
    bf16*  Qb   = (bf16*) take((size_t)BATCH * 512 * DM * 2);
    bf16*  Kb   = (bf16*) take((size_t)BATCH * 512 * DM * 2);
    bf16*  VT   = (bf16*) take((size_t)BATCH * 512 * DM * 2);   // per-batch transposed V
    bf16*  ATb  = (bf16*) take((size_t)BATCH * 512 * DM * 2);
    float* P32  = (float*)take((size_t)BATCH * 512 * DM * 4);
    float* O132 = (float*)take((size_t)BATCH * 512 * DM * 4);
    bf16*  O1bf = (bf16*) take((size_t)BATCH * 512 * DM * 2);
    bf16*  Hbf  = (bf16*) take((size_t)BATCH * 512 * FFD * 2);
    float* SC   = (float*)take((size_t)BATCH * 512 * 4);
    const size_t LW = 4 * 16384 + 2 * 65536;    // bf16 elems per layer
    bf16*  WB   = (bf16*) take((size_t)NLY * LW * 2);

    // convert GEMM weights to transposed bf16 once per launch (deterministic)
    for (int li = 0; li < NLY; ++li) {
        bf16* wb = WB + (size_t)li * LW;
        const int base = LBASE + li * NPARAM;
        const int which[6] = {0, 1, 2, 3, 7, 9};            // Wq Wk Wv Wo W1 W2
        const int woff[6]  = {0, 16384, 32768, 49152, 65536, 131072};
        const int wdi[6]   = {DM, DM, DM, DM, DM, FFD};     // Din
        const int wdo[6]   = {DM, DM, DM, DM, FFD, DM};     // Dout
        for (int j = 0; j < 6; ++j) {
            int n = wdi[j] * wdo[j];
            k_cvt_t<<<dim3((n + 255) / 256), 256, 0, stream>>>(
                (const float*)d_in[base + which[j]], wb + woff[j], wdi[j], wdo[j]);
        }
    }

    // embedding
    k_embed<<<dim3((BATCH * 512 * DM + 255) / 256), 256, 0, stream>>>(
        data, embW, embB, XA32, XAbf, BATCH * 512);

    auto encoder_layer = [&](float* Xin32, bf16* Xinbf, float* Xout32, bf16* Xoutbf,
                             int N, int li, const float* mask) {
        int R = BATCH * N;
        int logN = 31 - __builtin_clz((unsigned)N);
        const int base = LBASE + li * NPARAM;
        const float* bo  = (const float*)d_in[base + 4];
        const float* g1  = (const float*)d_in[base + 5];
        const float* be1 = (const float*)d_in[base + 6];
        const float* bf1 = (const float*)d_in[base + 8];
        const float* bf2 = (const float*)d_in[base + 10];
        const float* g2  = (const float*)d_in[base + 11];
        const float* be2 = (const float*)d_in[base + 12];
        bf16* wb = WB + (size_t)li * LW;
        bf16 *WqT = wb, *WkT = wb + 16384, *WvT = wb + 32768, *WoT = wb + 49152,
             *W1T = wb + 65536, *W2T = wb + 131072;

        dim3 gP(R / 16, DM / 32);
        k_gemm<false,false,false,false,true,false><<<gP, 32, 0, stream>>>(
            Xinbf, WqT, nullptr, nullptr, nullptr, Qb, nullptr, DM, DM, logN);
        k_gemm<false,false,false,false,true,false><<<gP, 32, 0, stream>>>(
            Xinbf, WkT, nullptr, nullptr, nullptr, Kb, nullptr, DM, DM, logN);
        k_gemm<false,false,false,false,false,true><<<gP, 32, 0, stream>>>(
            Xinbf, WvT, nullptr, nullptr, nullptr, nullptr, VT, DM, DM, logN);
        k_attn<<<dim3(N / 16, HN, BATCH), 32, (size_t)16 * N * sizeof(float), stream>>>(
            Qb, Kb, VT, mask, ATb, N);
        k_gemm<true,true,false,true,false,false><<<gP, 32, 0, stream>>>(
            ATb, WoT, bo, Xin32, P32, nullptr, nullptr, DM, DM, logN);
        k_inorm<<<dim3((BATCH * DM + 127) / 128), 128, 0, stream>>>(P32, g1, be1, O132, O1bf, N);
        k_gemm<true,false,true,false,true,false><<<dim3(R / 16, FFD / 32), 32, 0, stream>>>(
            O1bf, W1T, bf1, nullptr, nullptr, Hbf, nullptr, DM, FFD, logN);
        k_gemm<true,true,false,true,false,false><<<gP, 32, 0, stream>>>(
            Hbf, W2T, bf2, O132, P32, nullptr, nullptr, FFD, DM, logN);
        k_inorm<<<dim3((BATCH * DM + 127) / 128), 128, 0, stream>>>(P32, g2, be2, Xout32, Xoutbf, N);
    };

    int N = 512;
    float* curX = XA32; bf16* curXb = XAbf;
    float* altX = XB32; bf16* altXb = XBbf;
    const float* curMask = mask0;

    for (int blk = 0; blk < 3; ++blk) {
        for (int ly = 0; ly < 3; ++ly)
            encoder_layer(curX, curXb, curX, curXb, N, blk * 4 + ly, curMask);
        // score layer feeds only the scoring head; write its output into P32 in place
        encoder_layer(curX, curXb, P32, nullptr, N, blk * 4 + 3, curMask);
        int R = BATCH * N;
        k_score<<<dim3((R + 127) / 128), 128, 0, stream>>>(
            P32, (const float*)d_in[PBASE + 2 * blk], (const float*)d_in[PBASE + 2 * blk + 1],
            curMask, SC, R);
        int kk = N / 2;
        k_topk<<<dim3(BATCH), 32, 0, stream>>>(SC, curX, altX, altXb, N, kk);
        float* t = curX; curX = altX; altX = t;
        bf16* tb = curXb; curXb = altXb; altXb = tb;
        N = kk;
        curMask = nullptr;    // mask resets to zeros after each block
    }

    k_pool<<<dim3((BATCH * DM + 127) / 128), 128, 0, stream>>>(curX, (float*)d_out, N);
}
